// MoeFFN_27221502722583
// MI455X (gfx1250) — compile-verified, matrix operand order checked by reading, not run on hardware
//
#include <hip/hip_runtime.h>
#include <math.h>

// ---------------------------------------------------------------------------
// MoE FFN (top-2 of 8 experts), D=1024, H=4096, T=8192, bf16 WMMA path.
// ---------------------------------------------------------------------------

constexpr int kD = 1024;   // d_model
constexpr int kH = 4096;   // hidden
constexpr int kT = 8192;   // tokens (B*N)
constexpr int kE = 8;      // experts
constexpr int kHP = 2048;  // hidden per pass (H split into 2 passes for LDS)
constexpr int kM = 32;     // token rows per FFN block

typedef __attribute__((ext_vector_type(16))) __bf16 v16bf;
typedef __attribute__((ext_vector_type(8)))  __bf16 v8bf;
typedef __attribute__((ext_vector_type(8)))  float  v8f;
typedef __attribute__((ext_vector_type(4)))  int    v4i;

__device__ __forceinline__ v16bf bf_cat(v8bf lo, v8bf hi) {
  return __builtin_shufflevector(lo, hi, 0, 1, 2, 3, 4, 5, 6, 7,
                                 8, 9, 10, 11, 12, 13, 14, 15);
}

__device__ __forceinline__ v8bf cvt8_bf16(float4 a, float4 b) {
  v8bf r;
  r[0] = (__bf16)a.x; r[1] = (__bf16)a.y; r[2] = (__bf16)a.z; r[3] = (__bf16)a.w;
  r[4] = (__bf16)b.x; r[5] = (__bf16)b.y; r[6] = (__bf16)b.z; r[7] = (__bf16)b.w;
  return r;
}

// A operand (16-bit 16x32 layout): lane<16 holds K=kb..kb+7 and kb+16..kb+23
// with kb = 8*(lane>>4). Row-major bf16 source -> two contiguous 16B reads.
__device__ __forceinline__ v16bf load_a_lds(const __bf16* rowbase, int k) {
  v8bf lo = *(const v8bf*)(rowbase + k);
  v8bf hi = *(const v8bf*)(rowbase + k + 16);
  return bf_cat(lo, hi);
}

// B operand built identically from a row-major f32 weight row (C = A * W^T).
__device__ __forceinline__ v16bf load_b_global(const float* rowbase, int k) {
  const float4* p = (const float4*)(rowbase + k);
  const float4* q = (const float4*)(rowbase + k + 16);
  v8bf lo = cvt8_bf16(p[0], p[1]);
  v8bf hi = cvt8_bf16(q[0], q[1]);
  return bf_cat(lo, hi);
}

__device__ __forceinline__ v8f wmma_bf16(v16bf a, v16bf b, v8f c) {
  return __builtin_amdgcn_wmma_f32_16x16x32_bf16(false, a, false, b,
                                                 (short)0, c, false, false);
}

// Branch-free exact-erf GELU (Abramowitz-Stegun 7.1.26, |err| < 1.5e-7).
// Avoids libm erff's exec-mask branches between WMMA regions.
__device__ __forceinline__ float gelu_fast(float v) {
  const float z  = v * 0.70710678118654752f;   // x / sqrt(2)
  const float az = __builtin_fabsf(z);
  const float t  = __builtin_amdgcn_rcpf(__builtin_fmaf(0.3275911f, az, 1.0f));
  float p = __builtin_fmaf(t, 1.061405429f, -1.453152027f);
  p = __builtin_fmaf(t, p, 1.421413741f);
  p = __builtin_fmaf(t, p, -0.284496736f);
  p = __builtin_fmaf(t, p, 0.254829592f);
  // exp(-az^2) = exp2(-az^2 * log2(e)); v_exp_f32 is exp2 -> branch-free
  const float e   = __builtin_amdgcn_exp2f(az * az * -1.4426950408889634f);
  const float erf_abs = __builtin_fmaf(-p * t, e, 1.0f);
  const float erf_z   = (z < 0.0f) ? -erf_abs : erf_abs;
  return 0.5f * v * (1.0f + erf_z);
}

__device__ __forceinline__ float exp_fast(float v) {
  return __builtin_amdgcn_exp2f(v * 1.4426950408889634f);
}

__device__ __forceinline__ void atom_add_f32(float* p, float v) {
  __hip_atomic_fetch_add(p, v, __ATOMIC_RELAXED, __HIP_MEMORY_SCOPE_AGENT);
}

// ---- CDNA5 async global->LDS copy (ASYNCcnt path), with safe fallback ----
__device__ __forceinline__ void async_copy_b128(const float* gsrc, float* ldst) {
#if defined(__AMDGCN__) && __has_builtin(__builtin_amdgcn_global_load_async_to_lds_b128)
  __builtin_amdgcn_global_load_async_to_lds_b128(
      (__attribute__((address_space(1))) v4i*)(unsigned long long)(uintptr_t)gsrc,
      (__attribute__((address_space(3))) v4i*)(unsigned)(uintptr_t)ldst,
      0, 0);
#else
  *(float4*)ldst = *(const float4*)gsrc;
#endif
}

__device__ __forceinline__ void wait_asynccnt0() {
#if defined(__AMDGCN__) && __has_builtin(__builtin_amdgcn_s_wait_asynccnt)
  __builtin_amdgcn_s_wait_asynccnt(0);
#elif defined(__AMDGCN__)
  asm volatile("s_wait_asynccnt 0x0" ::: "memory");
#endif
}

// ---------------------------------------------------------------------------
// Kernel 0: zero output buffer + workspace counters (graph-capture safe).
// ---------------------------------------------------------------------------
__global__ void moe_init_kernel(float* __restrict__ out, int out_n,
                                int* __restrict__ cnt_gsum) {
  size_t i = (size_t)blockIdx.x * blockDim.x + threadIdx.x;
  size_t stride = (size_t)gridDim.x * blockDim.x;
  for (; i < (size_t)out_n; i += stride) out[i] = 0.0f;
  if (blockIdx.x == 0 && threadIdx.x < 16) cnt_gsum[threadIdx.x] = 0;
}

// ---------------------------------------------------------------------------
// Kernel 1: router. One wave32 per token.
// ---------------------------------------------------------------------------
__global__ void moe_router_kernel(const float* __restrict__ x,
                                  const float* __restrict__ Wr,
                                  int* __restrict__ cnt,
                                  float* __restrict__ gsum,
                                  int* __restrict__ tok,
                                  float* __restrict__ gwt) {
  const int lane = threadIdx.x & 31;
  const int wave = threadIdx.x >> 5;
  const int t = blockIdx.x * 8 + wave;   // T divisible by 8

  const float* xr = x + (size_t)t * kD;
  float acc[kE];
#pragma unroll
  for (int e = 0; e < kE; ++e) acc[e] = 0.0f;

  for (int i = 0; i < kD / 32; ++i) {
    float xv = xr[lane + 32 * i];
#pragma unroll
    for (int e = 0; e < kE; ++e) acc[e] += xv * Wr[e * kD + lane + 32 * i];
  }
#pragma unroll
  for (int e = 0; e < kE; ++e) {
#pragma unroll
    for (int off = 16; off > 0; off >>= 1) acc[e] += __shfl_xor(acc[e], off, 32);
  }

  // softmax over 8 logits (each lane redundantly)
  float mx = acc[0];
#pragma unroll
  for (int e = 1; e < kE; ++e) mx = fmaxf(mx, acc[e]);
  float g[kE];
  float s = 0.0f;
#pragma unroll
  for (int e = 0; e < kE; ++e) { g[e] = exp_fast(acc[e] - mx); s += g[e]; }
  float inv = 1.0f / s;
#pragma unroll
  for (int e = 0; e < kE; ++e) g[e] *= inv;

  // top-2 (first index wins ties, matching lax.top_k)
  int e1 = 0;
#pragma unroll
  for (int e = 1; e < kE; ++e) if (g[e] > g[e1]) e1 = e;
  int e2 = (e1 == 0) ? 1 : 0;
#pragma unroll
  for (int e = 0; e < kE; ++e)
    if (e != e1 && g[e] > g[e2]) e2 = e;

  float denom = fmaxf(g[e1] + g[e2], 1e-9f);
  float w1 = g[e1] / denom;
  float w2 = g[e2] / denom;

  if (lane == 0) {
    int s1 = atomicAdd(&cnt[e1], 1);
    tok[e1 * kT + s1] = t;
    gwt[e1 * kT + s1] = w1;
    int s2 = atomicAdd(&cnt[e2], 1);
    tok[e2 * kT + s2] = t;
    gwt[e2 * kT + s2] = w2;
#pragma unroll
    for (int e = 0; e < kE; ++e) atom_add_f32(&gsum[e], g[e]);
  }
}

// ---------------------------------------------------------------------------
// Kernel 2: aux loss finalize. aux = E * sum_e mean(gates_e) * mean(sel_e)
// ---------------------------------------------------------------------------
__global__ void moe_aux_kernel(const int* __restrict__ cnt,
                               const float* __restrict__ gsum,
                               float* __restrict__ aux_out) {
  if (threadIdx.x == 0 && blockIdx.x == 0) {
    const float invT = 1.0f / (float)kT;
    float a = 0.0f;
#pragma unroll
    for (int e = 0; e < kE; ++e)
      a += (gsum[e] * invT) * ((float)cnt[e] * invT);
    aux_out[0] = (float)kE * a;
  }
}

// ---------------------------------------------------------------------------
// Kernel 3: fused expert FFN.  Block = 512 threads (16 wave32s), 32 tokens.
//   stage:  X rows async-copied (ASYNCcnt) into LDS f32, converted to bf16
//   pass p in {0,1}: h[32,2048] = gelu(X[32,1024] @ W1_p^T + b1) in LDS (bf16)
//                    acc[32,1024] += h @ W2_p^T
//   final: out[token] += (acc + b2) * gate   (f32 global atomics)
// ---------------------------------------------------------------------------
__global__ void __launch_bounds__(512)
moe_ffn_kernel(const float* __restrict__ x,
               const float* __restrict__ W1, const float* __restrict__ b1,
               const float* __restrict__ W2, const float* __restrict__ b2,
               float* __restrict__ out,
               const int* __restrict__ cnt,
               const int* __restrict__ tok,
               const float* __restrict__ gwt) {
  const int e = blockIdx.y;
  const int count = cnt[e];
  if ((int)blockIdx.x * kM >= count) return;

  extern __shared__ char smem[];
  int*    sIds = (int*)smem;                          // 32 ints
  float*  sW   = (float*)(smem + 128);                // 32 floats
  __bf16* Xs   = (__bf16*)(smem + 256);               // [32][1024] bf16 (64KB)
  __bf16* hs   = (__bf16*)(smem + 256 + kM * kD * 2); // [32][2048] bf16 (128KB)
  float*  Xstg = (float*)hs;  // f32 staging for X, overlaps hs (exactly 128KB)

  const int lane = threadIdx.x & 31;
  const int wave = threadIdx.x >> 5;     // 0..15
  const int half = lane >> 4;            // 0/1
  const int nlo  = lane & 15;
  const int kb   = half * 8;

  // ---- stage token ids / gates ----
  if (threadIdx.x < kM) {
    int idx = blockIdx.x * kM + threadIdx.x;
    int t = 0;
    float w = 0.0f;
    if (idx < count) { t = tok[e * kT + idx]; w = gwt[e * kT + idx]; }
    sIds[threadIdx.x] = t;
    sW[threadIdx.x] = w;
  }
  __syncthreads();

  // ---- async-gather X rows (f32) into LDS staging ----
  for (int i = threadIdx.x; i < kM * (kD / 4); i += 512) {
    int row = i >> 8;            // kD/4 = 256 float4 per row
    int c4  = i & 255;
    async_copy_b128(x + (size_t)sIds[row] * kD + c4 * 4,
                    Xstg + (size_t)row * kD + c4 * 4);
  }
  wait_asynccnt0();
  __syncthreads();

  // ---- convert staged f32 -> bf16 A-tile ----
  for (int i = threadIdx.x; i < kM * (kD / 4); i += 512) {
    float4 v = *(const float4*)(Xstg + (size_t)i * 4);
    __bf16* dst = Xs + (size_t)i * 4;
    dst[0] = (__bf16)v.x; dst[1] = (__bf16)v.y;
    dst[2] = (__bf16)v.z; dst[3] = (__bf16)v.w;
  }
  __syncthreads();

  // persistent out accumulators: 4 D-colblocks x 2 rowblocks per wave
  v8f acc2[4][2];
#pragma unroll
  for (int c = 0; c < 4; ++c) {
    acc2[c][0] = (v8f)(0.0f);
    acc2[c][1] = (v8f)(0.0f);
  }

  for (int p = 0; p < 2; ++p) {
    // ================= GEMM1: h = gelu(X @ W1_p^T + b1) =================
    // 128 colblocks of 16 over this pass's 2048 hidden cols, 16 waves.
#pragma unroll 1
    for (int j = 0; j < 8; ++j) {
      const int cb   = wave + 16 * j;             // 0..127
      const int ncol = p * kHP + cb * 16 + nlo;   // W1 row / hidden col
      const float* w1row = W1 + ((size_t)e * kH + ncol) * kD;
      const __bf16* a0row = Xs + (size_t)nlo * kD;
      const __bf16* a1row = Xs + (size_t)(nlo + 16) * kD;

      v8f c0 = (v8f)(0.0f);
      v8f c1 = (v8f)(0.0f);
#pragma unroll 2
      for (int ks = 0; ks < kD / 32; ++ks) {
        const int k = ks * 32 + kb;
        v16bf bm = load_b_global(w1row, k);
        v16bf a0 = load_a_lds(a0row, k);
        v16bf a1 = load_a_lds(a1row, k);
        c0 = wmma_bf16(a0, bm, c0);
        c1 = wmma_bf16(a1, bm, c1);
      }
      const float b1v = b1[e * kH + ncol];
      const int colL = cb * 16 + nlo;
#pragma unroll
      for (int v = 0; v < 8; ++v) {
        int m0 = v + 8 * half;
        hs[(size_t)m0 * kHP + colL]        = (__bf16)gelu_fast(c0[v] + b1v);
        hs[(size_t)(m0 + 16) * kHP + colL] = (__bf16)gelu_fast(c1[v] + b1v);
      }
    }
    __syncthreads();

    // ================= GEMM2: acc += h @ W2_p^T =================
    const float* w2row[4];
#pragma unroll
    for (int c = 0; c < 4; ++c) {
      int n2 = (wave * 4 + c) * 16 + nlo;
      w2row[c] = W2 + ((size_t)e * kD + n2) * kH + p * kHP;
    }
    const __bf16* h0row = hs + (size_t)nlo * kHP;
    const __bf16* h1row = hs + (size_t)(nlo + 16) * kHP;
#pragma unroll 1
    for (int ks = 0; ks < kHP / 32; ++ks) {
      const int k = ks * 32 + kb;
      v16bf a0 = load_a_lds(h0row, k);
      v16bf a1 = load_a_lds(h1row, k);
#pragma unroll
      for (int c = 0; c < 4; ++c) {
        v16bf bm = load_b_global(w2row[c], k);
        acc2[c][0] = wmma_bf16(a0, bm, acc2[c][0]);
        acc2[c][1] = wmma_bf16(a1, bm, acc2[c][1]);
      }
    }
    __syncthreads();
  }

  // ---- epilogue: out[token] += (acc + b2) * gate ----
#pragma unroll
  for (int c = 0; c < 4; ++c) {
    const int n2 = (wave * 4 + c) * 16 + nlo;
    const float b2v = b2[e * kD + n2];
#pragma unroll
    for (int rb = 0; rb < 2; ++rb) {
#pragma unroll
      for (int v = 0; v < 8; ++v) {
        int m = v + 8 * half + 16 * rb;
        float val = (acc2[c][rb][v] + b2v) * sW[m];
        atom_add_f32(&out[(size_t)sIds[m] * kD + n2], val);
      }
    }
  }
}

// ---------------------------------------------------------------------------
// Host launcher.
// Workspace layout: [0)  int   cnt[8]
//                   [32) float gsum[8]
//                   [64) int   tok[8][8192]
//                   [64+256K) float gwt[8][8192]
// ---------------------------------------------------------------------------
extern "C" void kernel_launch(void* const* d_in, const int* in_sizes, int n_in,
                              void* d_out, int out_size, void* d_ws, size_t ws_size,
                              hipStream_t stream) {
  (void)in_sizes; (void)n_in; (void)ws_size;
  const float* x  = (const float*)d_in[0];
  const float* Wr = (const float*)d_in[1];
  const float* W1 = (const float*)d_in[2];
  const float* b1 = (const float*)d_in[3];
  const float* W2 = (const float*)d_in[4];
  const float* b2 = (const float*)d_in[5];
  float* out = (float*)d_out;

  int*   cnt  = (int*)d_ws;
  float* gsum = (float*)((char*)d_ws + 32);
  int*   tok  = (int*)((char*)d_ws + 64);
  float* gwt  = (float*)((char*)d_ws + 64 + (size_t)kE * kT * 4);

  moe_init_kernel<<<2048, 256, 0, stream>>>(out, out_size, cnt);
  moe_router_kernel<<<kT / 8, 256, 0, stream>>>(x, Wr, cnt, gsum, tok, gwt);
  moe_aux_kernel<<<1, 32, 0, stream>>>(cnt, gsum, out + (size_t)kT * kD);

  dim3 grid(kT / kM, kE);
  size_t shmem = 256 + (size_t)kM * kD * 2 + (size_t)kM * kHP * 2;  // 192.25 KB
  moe_ffn_kernel<<<grid, 512, shmem, stream>>>(x, W1, b1, W2, b2, out,
                                               cnt, tok, gwt);
}
